// MultiHeadSelfAttentionBlock_22402549416491
// MI455X (gfx1250) — compile-verified, matrix operand order checked by reading, not run on hardware
//
#include <hip/hip_runtime.h>
#include <hip/hip_bf16.h>
#include <math.h>

// MI455X / gfx1250: wave32, WMMA 16x16x32 f16 -> f32 accumulate.
// Pipeline: z@Wqkv (f32->f16 out) -> flash attention (f16, TDM-fed K/Q tiles)
//           -> attn@Wout (f32 out).

typedef __attribute__((ext_vector_type(16))) _Float16 v16h;
typedef __attribute__((ext_vector_type(8)))  float    v8f;
typedef __attribute__((ext_vector_type(4)))  unsigned int v4u;
typedef __attribute__((ext_vector_type(8)))  int      v8i;
typedef __attribute__((ext_vector_type(4)))  int      v4i;

#if __has_include(<hip/amd_detail/amd_gfx1250_TDM.h>)
#define TDM_6ARG 1
#else
#define TDM_6ARG 0
#endif

#define WMMA_F16(a, b, c) \
  __builtin_amdgcn_wmma_f32_16x16x32_f16(false, (a), false, (b), (short)0, (c), false, false)

// Load a 16x32 f16 A/B fragment from an LDS tile laid out [rows][stride halves].
// ISA layout (CDNA5 ISA): lane L holds row (L&15); kb = 8*(L>=16);
// VGPR0-3 = K {kb..kb+7}, VGPR4-7 = K {16+kb..16+kb+7}.
__device__ __forceinline__ v16h frag_ld(const _Float16* base, int stride, int row,
                                        int koff, int lane) {
  const int r  = lane & 15;
  const int kb = (lane >> 4) << 3;  // 0 or 8
  const _Float16* p = base + (size_t)(row + r) * stride + koff + kb;
  union { v16h h; uint4 u[2]; } f;
  f.u[0] = *(const uint4*)(p);
  f.u[1] = *(const uint4*)(p + 16);
  return f.h;
}

// ---------------------------------------------------------------------------
// Tensor Data Mover: 2D f16 tile -> LDS, with HW padding of 4 dwords every
// 32 dwords so a 64-half tile row lands with LDS stride 72 halves.
// D# layout per CDNA5 ISA.
// Issue from ONE wave; completion via s_wait_tensorcnt + workgroup barrier.
// ---------------------------------------------------------------------------
__device__ __forceinline__ void tdm_load_2d_f16(const void* gaddr, unsigned lds_addr,
                                                unsigned tile_d0, unsigned tile_d1,
                                                unsigned tensor_d0, unsigned tensor_d1,
                                                unsigned row_stride_elems) {
  unsigned long long ga = (unsigned long long)(uintptr_t)gaddr;
  v4u g0;
  g0.x = 1u;                                   // count=1, user descriptor
  g0.y = lds_addr;                             // LDS byte address
  g0.z = (unsigned)ga;                         // global_addr[31:0]
  g0.w = (unsigned)(ga >> 32) | (2u << 30);    // global_addr[56:32] | type=2
  v8i g1;
  // data_size=1 (2B), pad_enable, pad_interval=4 (32 dw), pad_amount=3 (4 dw)
  g1[0] = (int)((1u << 16) | (1u << 20) | (4u << 22) | (3u << 25));
  g1[1] = (int)(tensor_d0 << 16);                              // tensor_dim0 lo
  g1[2] = (int)((tensor_d0 >> 16) | (tensor_d1 << 16));        // dim0 hi | dim1 lo
  g1[3] = (int)((tensor_d1 >> 16) | (tile_d0 << 16));          // dim1 hi | tile_dim0
  g1[4] = (int)tile_d1;                                        // tile_dim1 (tile_dim2=0)
  g1[5] = (int)row_stride_elems;                               // tensor_dim0_stride lo
  g1[6] = 0;
  g1[7] = 0;
  v4i z4 = {0, 0, 0, 0};
#if TDM_6ARG
  v8i z8 = {0, 0, 0, 0, 0, 0, 0, 0};
  __builtin_amdgcn_tensor_load_to_lds(g0, g1, z4, z4, z8, 0);
#else
  __builtin_amdgcn_tensor_load_to_lds(g0, g1, z4, z4, 0);
#endif
}

// ---------------------------------------------------------------------------
// GEMM: C[M,N] = A[M,K] * B[K,N]; A is f32 or f16, B is f32, C is f32 or f16.
// Block tile 128x128, K-tile 64; 256 threads = 8 waves as 4(M) x 2(N);
// wave tile 32x64 -> 16 WMMAs per barrier phase.
// ---------------------------------------------------------------------------
template <typename TA, typename TO>
__global__ __launch_bounds__(256) void gemm_wmma_kernel(
    const TA* __restrict__ A, const float* __restrict__ B, TO* __restrict__ C,
    int M, int N, int K) {
  __shared__ _Float16 sA[128 * 72];  // [row][k0..63], stride 72 halves
  __shared__ _Float16 sB[128 * 72];  // [col][k0..63] (K-transposed)

  const int tid  = threadIdx.x;
  const int lane = tid & 31;
  const int wave = tid >> 5;
  const int wm   = (wave & 3) * 32;
  const int wn   = (wave >> 2) * 64;
  const int bm   = blockIdx.x * 128;
  const int bn   = blockIdx.y * 128;

  const v8f vzero = {0.f, 0.f, 0.f, 0.f, 0.f, 0.f, 0.f, 0.f};
  v8f acc[2][4];
#pragma unroll
  for (int i = 0; i < 2; ++i)
#pragma unroll
    for (int j = 0; j < 4; ++j) acc[i][j] = vzero;

  for (int k0 = 0; k0 < K; k0 += 64) {
    // Prefetch next K-tile into cache (global_prefetch_b8).
    if (k0 + 64 < K) {
      __builtin_prefetch(A + (size_t)(bm + (tid >> 4)) * K + k0 + 64 + (tid & 15) * 4, 0, 1);
      __builtin_prefetch(B + (size_t)(k0 + 64 + (tid >> 5) * 2) * N + bn + (tid & 31) * 4, 0, 1);
    }
    // A tile: 128 rows x 64 k, 8 groups of 4 elems per thread.
#pragma unroll
    for (int i = 0; i < 8; ++i) {
      int g = tid + i * 256;
      int row = g >> 4;
      int c4 = (g & 15) * 4;
      if constexpr (sizeof(TA) == 4) {
        const float4 ld = *(const float4*)(A + (size_t)(bm + row) * K + k0 + c4);
        union { _Float16 h[4]; uint2 u; } t;
        t.h[0] = (_Float16)ld.x; t.h[1] = (_Float16)ld.y;
        t.h[2] = (_Float16)ld.z; t.h[3] = (_Float16)ld.w;
        *(uint2*)&sA[row * 72 + c4] = t.u;
      } else {
        *(uint2*)&sA[row * 72 + c4] = *(const uint2*)(A + (size_t)(bm + row) * K + k0 + c4);
      }
    }
    // B tile: 64 k x 128 n; read two consecutive k rows and store packed b32
    // pairs into the K-transposed [col][k] layout.
#pragma unroll
    for (int i = 0; i < 4; ++i) {
      int g = tid + i * 256;
      int kp = g >> 5;            // k pair 0..31
      int c4 = (g & 31) * 4;      // col group
      const float4 la = *(const float4*)(B + (size_t)(k0 + 2 * kp) * N + bn + c4);
      const float4 lb = *(const float4*)(B + (size_t)(k0 + 2 * kp + 1) * N + bn + c4);
      const float a[4] = {la.x, la.y, la.z, la.w};
      const float b[4] = {lb.x, lb.y, lb.z, lb.w};
#pragma unroll
      for (int j = 0; j < 4; ++j) {
        union { _Float16 h[2]; unsigned u; } t;
        t.h[0] = (_Float16)a[j];
        t.h[1] = (_Float16)b[j];
        *(unsigned*)&sB[(c4 + j) * 72 + 2 * kp] = t.u;
      }
    }
    __syncthreads();

#pragma unroll
    for (int kc = 0; kc < 2; ++kc) {
      v16h af[2], bf[4];
#pragma unroll
      for (int mi = 0; mi < 2; ++mi) af[mi] = frag_ld(sA, 72, wm + mi * 16, kc * 32, lane);
#pragma unroll
      for (int ni = 0; ni < 4; ++ni) bf[ni] = frag_ld(sB, 72, wn + ni * 16, kc * 32, lane);
#pragma unroll
      for (int mi = 0; mi < 2; ++mi)
#pragma unroll
        for (int ni = 0; ni < 4; ++ni)
          acc[mi][ni] = WMMA_F16(af[mi], bf[ni], acc[mi][ni]);
    }
    __syncthreads();
  }

  // C/D layout: VGPR r = row r (lanes 0-15) / r+8 (lanes 16-31); col = lane&15.
  const int hi = (lane >> 4) * 8;
  const int cl = lane & 15;
#pragma unroll
  for (int mi = 0; mi < 2; ++mi)
#pragma unroll
    for (int ni = 0; ni < 4; ++ni)
#pragma unroll
      for (int r = 0; r < 8; ++r) {
        int row = bm + wm + mi * 16 + r + hi;
        int col = bn + wn + ni * 16 + cl;
        C[(size_t)row * N + col] = (TO)acc[mi][ni][r];
      }
}

// ---------------------------------------------------------------------------
// Flash attention, causal. qkv: f16 [N, T, 3*D]; out: f16 [N, T, D].
// Grid: (T/128, H, N). 256 threads = 8 waves; wave w owns 16 query rows.
// Q and K tiles are fetched by the Tensor Data Mover (wave 0 issues,
// s_wait_tensorcnt + barrier publishes); V is transposed manually into LDS.
// ---------------------------------------------------------------------------
__global__ __launch_bounds__(256) void flash_attn_kernel(
    const _Float16* __restrict__ qkv, _Float16* __restrict__ out) {
  constexpr int T = 2048, D = 1024, D3 = 3072, DK = 64;
  const int qb   = blockIdx.x * 128;
  const int h    = blockIdx.y;
  const int n    = blockIdx.z;
  const int tid  = threadIdx.x;
  const int lane = tid & 31;
  const int wave = tid >> 5;

  const _Float16* qp = qkv + (size_t)n * T * D3 + h * DK;
  const _Float16* kp = qp + D;
  const _Float16* vp = qp + 2 * D;

  __shared__ _Float16 sQ[128 * 72];    // [query][dk]   (TDM-filled, stride 72)
  __shared__ _Float16 sK[64 * 72];     // [key][dk]     (TDM-filled, stride 72)
  __shared__ _Float16 sV[64 * 72];     // [dk][key]     (manual transpose)
  __shared__ _Float16 sP[8][16 * 72];  // per-wave P staging (C-layout -> A-layout)

  // Q tile 128x64 via TDM.
  if (wave == 0) {
    tdm_load_2d_f16(qp + (size_t)qb * D3, (unsigned)(uintptr_t)(void*)sQ,
                    64, 128, 64, (unsigned)T, (unsigned)D3);
    __builtin_amdgcn_s_wait_tensorcnt(0);
  }
  __syncthreads();

  v16h qf[2];
  qf[0] = frag_ld(sQ, 72, wave * 16, 0, lane);
  qf[1] = frag_ld(sQ, 72, wave * 16, 32, lane);

  const v8f vzero = {0.f, 0.f, 0.f, 0.f, 0.f, 0.f, 0.f, 0.f};
  v8f accO[4];
#pragma unroll
  for (int i = 0; i < 4; ++i) accO[i] = vzero;
  float m_r[8], l_r[8];
#pragma unroll
  for (int r = 0; r < 8; ++r) { m_r[r] = -1e30f; l_r[r] = 0.f; }

  const int hi   = (lane >> 4) * 8;
  const int cl   = lane & 15;
  const int row0 = qb + wave * 16 + hi;  // + r gives the global query row
  const int nkb  = qb / 64 + 2;          // causal: key blocks covering cols <= qb+127

  for (int kblk = 0; kblk < nkb; ++kblk) {
    // K tile 64x64 via TDM (overlaps with the manual V transpose below).
    if (wave == 0) {
      tdm_load_2d_f16(kp + (size_t)kblk * 64 * D3, (unsigned)(uintptr_t)(void*)sK,
                      64, 64, 64, (unsigned)T, (unsigned)D3);
    }
    // V tile 64x64 stored transposed [dk][key], packed b32 stores (key pairs).
    {
      int kpair = tid >> 3;          // 0..31 -> keys 2kp, 2kp+1
      int d8 = (tid & 7) * 8;        // dim group
      union { uint4 u; _Float16 h[8]; } a, b;
      a.u = *(const uint4*)(vp + (size_t)(kblk * 64 + 2 * kpair) * D3 + d8);
      b.u = *(const uint4*)(vp + (size_t)(kblk * 64 + 2 * kpair + 1) * D3 + d8);
#pragma unroll
      for (int j = 0; j < 8; ++j) {
        union { _Float16 h[2]; unsigned u; } t;
        t.h[0] = a.h[j];
        t.h[1] = b.h[j];
        *(unsigned*)&sV[(d8 + j) * 72 + 2 * kpair] = t.u;
      }
    }
    if (wave == 0) __builtin_amdgcn_s_wait_tensorcnt(0);
    __syncthreads();

    // S = Q @ K^T (16 rows x 64 keys per wave).
    v8f s[4];
#pragma unroll
    for (int ni = 0; ni < 4; ++ni) {
      s[ni] = vzero;
      v16h kf0 = frag_ld(sK, 72, ni * 16, 0, lane);
      v16h kf1 = frag_ld(sK, 72, ni * 16, 32, lane);
      s[ni] = WMMA_F16(qf[0], kf0, s[ni]);
      s[ni] = WMMA_F16(qf[1], kf1, s[ni]);
    }

    // Scale + causal mask.
    float p[4][8];
#pragma unroll
    for (int ni = 0; ni < 4; ++ni)
#pragma unroll
      for (int r = 0; r < 8; ++r) {
        float v = s[ni][r] * 0.125f;  // DK^-0.5 = 1/8
        int kcol = kblk * 64 + ni * 16 + cl;
        p[ni][r] = (kcol <= row0 + r) ? v : -1e30f;
      }

    // Online softmax; row r spans the 16 lanes of a half-wave -> width-16
    // xor-shuffle reductions.
#pragma unroll
    for (int r = 0; r < 8; ++r) {
      float mx = fmaxf(fmaxf(p[0][r], p[1][r]), fmaxf(p[2][r], p[3][r]));
#pragma unroll
      for (int off = 8; off >= 1; off >>= 1) mx = fmaxf(mx, __shfl_xor(mx, off, 32));
      float m_new = fmaxf(m_r[r], mx);
      float alpha = __expf(m_r[r] - m_new);
      float sum = 0.f;
#pragma unroll
      for (int ni = 0; ni < 4; ++ni) {
        float e = __expf(p[ni][r] - m_new);
        p[ni][r] = e;
        sum += e;
      }
#pragma unroll
      for (int off = 8; off >= 1; off >>= 1) sum += __shfl_xor(sum, off, 32);
      l_r[r] = l_r[r] * alpha + sum;
      m_r[r] = m_new;
#pragma unroll
      for (int ni = 0; ni < 4; ++ni) accO[ni][r] *= alpha;
    }

    // Stage P through per-wave LDS to convert C-layout -> A-layout.
    _Float16* pw = &sP[wave][0];
#pragma unroll
    for (int ni = 0; ni < 4; ++ni)
#pragma unroll
      for (int r = 0; r < 8; ++r)
        pw[(r + hi) * 72 + ni * 16 + cl] = (_Float16)p[ni][r];

    // O += P @ V.
#pragma unroll
    for (int kc = 0; kc < 2; ++kc) {
      v16h pf = frag_ld(pw, 72, 0, kc * 32, lane);
#pragma unroll
      for (int ni = 0; ni < 4; ++ni) {
        v16h vf = frag_ld(sV, 72, ni * 16, kc * 32, lane);
        accO[ni] = WMMA_F16(pf, vf, accO[ni]);
      }
    }
    __syncthreads();
  }

  // Normalize and write [n][row][h*DK + col] as f16.
#pragma unroll
  for (int r = 0; r < 8; ++r) {
    float inv = 1.f / l_r[r];
    int row = row0 + r;
#pragma unroll
    for (int ni = 0; ni < 4; ++ni)
      out[(size_t)n * T * D + (size_t)row * D + h * DK + ni * 16 + cl] =
          (_Float16)(accO[ni][r] * inv);
  }
}

// ---------------------------------------------------------------------------
extern "C" void kernel_launch(void* const* d_in, const int* in_sizes, int n_in,
                              void* d_out, int out_size, void* d_ws, size_t ws_size,
                              hipStream_t stream) {
  constexpr int N = 4, T = 2048, D = 1024, H = 16;
  constexpr int M = N * T;  // 8192

  const float* z    = (const float*)d_in[0];  // [N,T,D]
  const float* Wqkv = (const float*)d_in[1];  // [D,3D]
  const float* Wout = (const float*)d_in[2];  // [D,D]
  float* out = (float*)d_out;                 // [N,T,D]

  _Float16* qkv  = (_Float16*)d_ws;                                   // [M,3D] f16
  _Float16* attn = (_Float16*)((char*)d_ws + (size_t)M * 3 * D * 2);  // [M,D] f16

  // 1) qkv = z @ Wqkv  (f32 in, f16 out)
  gemm_wmma_kernel<float, _Float16>
      <<<dim3(M / 128, (3 * D) / 128), 256, 0, stream>>>(z, Wqkv, qkv, M, 3 * D, D);

  // 2) causal flash attention per (n, h)
  flash_attn_kernel<<<dim3(T / 128, H, N), 256, 0, stream>>>(qkv, attn);

  // 3) out = attn @ Wout  (f16 in, f32 out)
  gemm_wmma_kernel<_Float16, float>
      <<<dim3(M / 128, D / 128), 256, 0, stream>>>(attn, Wout, out, M, D, D);
}
